// BetterCrossCoder_3831110828648
// MI455X (gfx1250) — matrix operand order repaired and checked by reading, not compiled, not run on hardware
//
#include <hip/hip_runtime.h>

// ---------------------------------------------------------------------------
// BetterCrossCoder for MI455X (gfx1250, wave32):
//   h   = x @ enc + enc_bias          -> bf16 WMMA (v_wmma_f32_16x16x32_bf16)
//   lat = topk32(h)                   -> block-wide iterative argmax
//   out = lat @ dec + dec_bias        -> 32-row gather GEMV (sparse decode)
// Workspace: h = [2048 x 32768] fp32 = 256 MB in d_ws.
// ---------------------------------------------------------------------------

typedef __attribute__((ext_vector_type(16))) __bf16       v16bf;
typedef __attribute__((ext_vector_type(2)))  __bf16       bf16x2;
typedef __attribute__((ext_vector_type(8)))  float        v8f;
typedef __attribute__((ext_vector_type(4)))  float        f32x4;
typedef __attribute__((ext_vector_type(2)))  float        f32x2;
typedef __attribute__((ext_vector_type(4)))  unsigned int u32x4;

constexpr int NBATCH = 2048;   // B
constexpr int DIM    = 2048;   // D (K of encode, N of decode)
constexpr int HID    = 32768;  // H
constexpr int KTOP   = 32;

constexpr int AS = 40;         // LDS row stride (bf16 elems) for A tile (bank-conflict pad)
constexpr int BS = 40;         // LDS row stride (bf16 elems) for B^T tile

// Single v_cvt_pk_bf16_f32 per pair.
__device__ __forceinline__ unsigned pack_bf16(float a, float b) {
  union { bf16x2 h; unsigned u; } p;
  p.h = __builtin_convertvector((f32x2){a, b}, bf16x2);
  return p.u;
}

union V16 { v16bf v; u32x4 u[2]; };

// ===========================================================================
// Kernel 1: h[m][n] = sum_k x[m][k]*enc[k][n] + bias[n]
// Block tile: M=64, N=128. 256 threads = 8 waves; wave w owns N-slice w*16,
// and 4 M-subtiles (4 accumulators). K chunked by 32, double-buffered LDS:
// global loads for chunk k+1 issue before chunk k's WMMAs (1 barrier/iter).
// All 5 LDS operand loads issue as a burst -> single s_wait_dscnt, then 4
// back-to-back WMMAs.
// ===========================================================================
__global__ __launch_bounds__(256) void encode_wmma_kernel(
    const float* __restrict__ x,
    const float* __restrict__ enc_a, const float* __restrict__ enc_b,
    const float* __restrict__ ebias_a, const float* __restrict__ ebias_b,
    const int*   __restrict__ in_model,
    float*       __restrict__ h)
{
  __shared__ __bf16 Alds[2][64 * AS];    // [m][k] bf16
  __shared__ __bf16 Blds[2][128 * BS];   // [n][k] bf16 (transposed enc chunk)

  const int    sel  = in_model[0];
  const float* enc  = sel ? enc_b : enc_a;
  const float* bias = sel ? ebias_b : ebias_a;

  const int t    = threadIdx.x;
  const int wave = t >> 5;
  const int lane = t & 31;
  const int n0   = blockIdx.x * 128;  // N tile base
  const int m0   = blockIdx.y * 64;   // M tile base

  v8f acc[4] = {};

  // staging assignments
  const int a_row = t >> 2;           // 0..63
  const int a_col = (t & 3) * 8;      // 0,8,16,24  (k within chunk)
  const int kp    = t >> 4;           // 0..15 -> k pair (2*kp, 2*kp+1)
  const int nb    = (t & 15) * 8;     // 0..120 (n within tile)

  const float* aptr  = x   + (size_t)(m0 + a_row) * DIM + a_col;
  const float* bptr0 = enc + (size_t)(2 * kp)     * HID + n0 + nb;
  const float* bptr1 = enc + (size_t)(2 * kp + 1) * HID + n0 + nb;

  // wave-level operand addressing (contiguous k = 16*(lane/16) + s, same for A and B)
  const int l16   = lane & 15;
  const int nlane = wave * 16 + l16;                // B: N = lane%16 within slice
  const int bko   = (lane < 16) ? 0 : 16;          // B: low lanes K0..15, high K16..31
  const int ako0  = (lane < 16) ? 0 : 16;          // A: low lanes K0..7,  high K16..23
  const int ako1  = ako0 + 8;                      // A: +8 for K8..15 / K24..31

  // staged-chunk registers
  f32x4 ax0, ax1, b0a, b0b, b1a, b1b;

  auto load_chunk = [&](int kc) {
    const f32x4* pa = (const f32x4*)(aptr + kc);
    ax0 = pa[0]; ax1 = pa[1];
    const f32x4* p0 = (const f32x4*)(bptr0 + (size_t)kc * HID);
    const f32x4* p1 = (const f32x4*)(bptr1 + (size_t)kc * HID);
    b0a = p0[0]; b0b = p0[1];
    b1a = p1[0]; b1b = p1[1];
  };
  auto store_chunk = [&](int buf) {
    unsigned* da = (unsigned*)&Alds[buf][a_row * AS + a_col];
    da[0] = pack_bf16(ax0[0], ax0[1]);
    da[1] = pack_bf16(ax0[2], ax0[3]);
    da[2] = pack_bf16(ax1[0], ax1[1]);
    da[3] = pack_bf16(ax1[2], ax1[3]);
    #pragma unroll
    for (int j = 0; j < 4; ++j) {
      *(unsigned*)&Blds[buf][(nb + j    ) * BS + 2 * kp] = pack_bf16(b0a[j], b1a[j]);
      *(unsigned*)&Blds[buf][(nb + j + 4) * BS + 2 * kp] = pack_bf16(b0b[j], b1b[j]);
    }
  };

  // prologue: stage chunk 0 into buffer 0
  load_chunk(0);
  store_chunk(0);

  for (int kc = 0, it = 0; kc < DIM; kc += 32, ++it) {
    __syncthreads();                       // buf cur ready; prev compute done
    const int  cur  = it & 1;
    const bool more = (kc + 32) < DIM;
    if (more) load_chunk(kc + 32);         // issue global loads early

    // ---- load ALL operands first (ds_load burst, one dscnt wait) ----
    V16 bop, aop[4];
    bop.u[0] = *(const u32x4*)&Blds[cur][nlane * BS + bko];
    bop.u[1] = *(const u32x4*)&Blds[cur][nlane * BS + bko + 8];
    #pragma unroll
    for (int sm = 0; sm < 4; ++sm) {
      const int m = sm * 16 + l16;
      aop[sm].u[0] = *(const u32x4*)&Alds[cur][m * AS + ako0];
      aop[sm].u[1] = *(const u32x4*)&Alds[cur][m * AS + ako1];
    }
    // ---- 4 back-to-back WMMAs ----
    #pragma unroll
    for (int sm = 0; sm < 4; ++sm) {
      acc[sm] = __builtin_amdgcn_wmma_f32_16x16x32_bf16(
          false, aop[sm].v, false, bop.v, (short)0, acc[sm], false, false);
    }

    if (more) store_chunk(cur ^ 1);        // other buffer: no barrier needed here
  }

  // ---- epilogue: add bias, store fp32 h tile.
  // C/D layout: VGPR r -> lanes 0-15: M=r, N=lane; lanes 16-31: M=8+r, N=lane-16.
  const int   ncol  = n0 + wave * 16 + l16;
  const float bn    = bias[ncol];
  const int   mbase = m0 + ((lane < 16) ? 0 : 8);
  #pragma unroll
  for (int sm = 0; sm < 4; ++sm) {
    #pragma unroll
    for (int r = 0; r < 8; ++r) {
      h[(size_t)(mbase + sm * 16 + r) * HID + ncol] = acc[sm][r] + bn;
    }
  }
}

// ===========================================================================
// Kernel 2: per-row top-32 (exact, lowest-index tie-break like lax.top_k),
// then sparse decode: out[row] = sum_k v_k * dec[i_k][:] + dec_bias.
// One block (256 threads) per row. Thread t owns 32 groups of 4 contiguous
// elements: idx = i*1024 + t*4 + j  (coalesced b128 scans).
// ===========================================================================
__global__ __launch_bounds__(256) void topk_decode_kernel(
    const float* __restrict__ h,
    const float* __restrict__ dec_a, const float* __restrict__ dec_b,
    const float* __restrict__ dbias_a, const float* __restrict__ dbias_b,
    const int*   __restrict__ out_model,
    float*       __restrict__ out)
{
  const int row = blockIdx.x;
  const int t   = threadIdx.x;
  const int sel = out_model[0];
  const float* dec   = sel ? dec_b : dec_a;
  const float* dbias = sel ? dbias_b : dbias_a;

  __shared__ float s_val[256];
  __shared__ int   s_idx[256];
  __shared__ float topv[KTOP];
  __shared__ int   topi[KTOP];

  const float* hrow = h + (size_t)row * HID;

  unsigned long long ex0 = 0ull, ex1 = 0ull;  // 128-bit exclusion mask
  float mval;
  int   midx;

  auto rescan = [&]() {
    mval = -__builtin_inff();
    midx = t * 4;
    #pragma unroll 4
    for (int i = 0; i < 32; ++i) {
      f32x4 v = *(const f32x4*)(hrow + (i << 10) + (t << 2));
      #pragma unroll
      for (int j = 0; j < 4; ++j) {
        const int bit = i * 4 + j;
        bool excl = (bit < 64) ? ((ex0 >> bit) & 1) : ((ex1 >> (bit - 64)) & 1);
        if (!excl && v[j] > mval) { mval = v[j]; midx = (i << 10) + (t << 2) + j; }
      }
    }
  };
  rescan();

  for (int k = 0; k < KTOP; ++k) {
    s_val[t] = mval; s_idx[t] = midx;
    __syncthreads();
    // tree argmax: larger value wins; on tie, smaller index wins
    for (int s = 128; s > 0; s >>= 1) {
      if (t < s) {
        float vo = s_val[t + s]; int io = s_idx[t + s];
        if (vo > s_val[t] || (vo == s_val[t] && io < s_idx[t])) {
          s_val[t] = vo; s_idx[t] = io;
        }
      }
      __syncthreads();
    }
    if (t == 0) { topv[k] = s_val[0]; topi[k] = s_idx[0]; }
    __syncthreads();
    const int wi = topi[k];
    if (((wi >> 2) & 255) == t) {   // winning thread: exclude + rescan slice
      const int bit = ((wi >> 10) << 2) | (wi & 3);
      if (bit < 64) ex0 |= 1ull << bit; else ex1 |= 1ull << (bit - 64);
      rescan();
    }
    __syncthreads();
  }

  // ---- sparse decode: each thread produces 8 output columns
  const int c0 = t * 8;
  float acc[8];
  #pragma unroll
  for (int j = 0; j < 8; ++j) acc[j] = dbias[c0 + j];
  for (int k = 0; k < KTOP; ++k) {
    const float  v    = topv[k];
    const float* drow = dec + (size_t)topi[k] * DIM + c0;
    f32x4 d0 = *(const f32x4*)(drow);
    f32x4 d1 = *(const f32x4*)(drow + 4);
    acc[0] += v * d0[0]; acc[1] += v * d0[1];
    acc[2] += v * d0[2]; acc[3] += v * d0[3];
    acc[4] += v * d1[0]; acc[5] += v * d1[1];
    acc[6] += v * d1[2]; acc[7] += v * d1[3];
  }
  #pragma unroll
  for (int j = 0; j < 8; ++j) out[(size_t)row * DIM + c0 + j] = acc[j];
}

// ===========================================================================
extern "C" void kernel_launch(void* const* d_in, const int* in_sizes, int n_in,
                              void* d_out, int out_size, void* d_ws, size_t ws_size,
                              hipStream_t stream) {
  const float* x        = (const float*)d_in[0];
  const float* enc_a    = (const float*)d_in[1];
  const float* enc_ab   = (const float*)d_in[2];
  const float* dec_a    = (const float*)d_in[3];
  const float* dec_ab   = (const float*)d_in[4];
  const float* enc_b    = (const float*)d_in[5];
  const float* enc_bb   = (const float*)d_in[6];
  const float* dec_b    = (const float*)d_in[7];
  const float* dec_bb   = (const float*)d_in[8];
  const int*   in_model  = (const int*)d_in[9];
  const int*   out_model = (const int*)d_in[10];
  float*       outp     = (float*)d_out;
  float*       h        = (float*)d_ws;   // needs NBATCH*HID*4 = 256 MB

  dim3 g1(HID / 128, NBATCH / 64);
  encode_wmma_kernel<<<g1, 256, 0, stream>>>(x, enc_a, enc_b, enc_ab, enc_bb,
                                             in_model, h);
  topk_decode_kernel<<<NBATCH, 256, 0, stream>>>(h, dec_a, dec_b, dec_ab, dec_bb,
                                                 out_model, outp);
}
